// NMultiheadSelfAttention_41008347742971
// MI455X (gfx1250) — matrix operand order repaired and checked by reading, not compiled, not run on hardware
//
#include <hip/hip_runtime.h>
#include <hip/hip_bf16.h>

// Problem constants (from reference): B=4, S=1024, E=1024, H=16, HD=64
#define B_  4
#define S_  1024
#define E_  1024
#define H_  16
#define HD_ 64
#define BH_ (B_ * H_)    // 64
#define M_  (B_ * S_)    // 4096 rows for the projection GEMMs

// Toggle the CDNA5 async-LDS staging path in scores_kernel (inline asm:
// global_load_async_to_lds_b128 + s_wait_asynccnt). Set to 0 to fall back to
// direct global fragment loads if the assembler rejects the mnemonics.
#define USE_ASYNC_LDS 1

typedef __attribute__((ext_vector_type(16))) __bf16 v16bf;
typedef __attribute__((ext_vector_type(8)))  __bf16 v8bf;
typedef __attribute__((ext_vector_type(8)))  float  v8f;

// ---------------------------------------------------------------------------
// CDNA5 WMMA fragment helpers (wave32, 16x16x32 bf16 -> f32)
//
// A fragment (16x32, M x K):
//   lanes 0-15 : row M = lane,     K = k0+0..15  (elems 0-15)
//   lanes16-31 : row M = lane-16,  K = k0+16..31 (elems 0-15)
// B fragment (32x16, K x N): lane%16 = N column; lanes 0-15 hold K=k0+0..15,
//   lanes 16-31 hold K=k0+16..31 (16 contiguous bf16 per lane).
// C/D fragment (16x16 f32): VGPR r: lanes 0-15 -> (M=r, N=lane),
//   lanes 16-31 -> (M=8+r, N=lane-16).
// ---------------------------------------------------------------------------

static __device__ __forceinline__ v16bf frag_a_bf16(const __bf16* __restrict__ base,
                                                    int ld, int m0, int k0, int lane) {
  const __bf16* p = base + (size_t)(m0 + (lane & 15)) * ld + k0 + ((lane & 16) ? 16 : 0);
  v8bf lo = *(const v8bf*)p;         // K chunk +0..7
  v8bf hi = *(const v8bf*)(p + 8);   // K chunk +8..15
  return __builtin_shufflevector(lo, hi, 0,1,2,3,4,5,6,7,8,9,10,11,12,13,14,15);
}

// A fragment gathered from an fp32 row-major source (convert to bf16 on load)
static __device__ __forceinline__ v16bf frag_a_f32(const float* __restrict__ base,
                                                   int ld, int m0, int k0, int lane) {
  const float* p = base + (size_t)(m0 + (lane & 15)) * ld + k0 + ((lane & 16) ? 16 : 0);
  v16bf f;
#pragma unroll
  for (int i = 0; i < 8; ++i) {
    f[i]     = (__bf16)p[i];
    f[i + 8] = (__bf16)p[i + 8];
  }
  return f;
}

// B fragment where the SOURCE is row-major [N][K] (i.e. we compute A * src^T).
// Each lane reads 16 contiguous bf16 (32 bytes) from one source row.
static __device__ __forceinline__ v16bf frag_b_rowmajorT(const __bf16* __restrict__ base,
                                                         int ld, int n0, int k0, int lane) {
  const __bf16* p = base + (size_t)(n0 + (lane & 15)) * ld + k0 + ((lane & 16) ? 16 : 0);
  return *(const v16bf*)p;
}

// Same pattern, reading a staged 16xHD tile from LDS (generic ptr -> ds_load)
static __device__ __forceinline__ v16bf frag_b_lds(const __bf16* tile, int k0, int lane) {
  const __bf16* p = tile + (lane & 15) * HD_ + k0 + ((lane & 16) ? 16 : 0);
  return *(const v16bf*)p;
}

static __device__ __forceinline__ v8f wmma_bf16(v16bf a, v16bf b, v8f c) {
  // 8 args: (neg_a, A, neg_b, B, c_mod, C, reuse_a, reuse_b)
  return __builtin_amdgcn_wmma_f32_16x16x32_bf16(false, a, false, b, (short)0, c,
                                                 false, false);
}

#if USE_ASYNC_LDS
// CDNA5 async load: GLOBAL_LOAD_ASYNC_TO_LDS_B128 (GV mode).
// VDST = VGPR holding the LDS byte offset, VADDR = 64-bit global address.
// Tracked with ASYNCcnt (ISA 10.x / 15.18.3).
static __device__ __forceinline__ void async_ld_b128(unsigned lds_off,
                                                     unsigned long long gaddr) {
  asm volatile("global_load_async_to_lds_b128 %0, %1, off"
               :: "v"(lds_off), "v"(gaddr) : "memory");
}
static __device__ __forceinline__ void wait_async0() {
  asm volatile("s_wait_asynccnt 0x0" ::: "memory");
}
static __device__ __forceinline__ void wait_async1() {
  asm volatile("s_wait_asynccnt 0x1" ::: "memory");
}
#endif

// ---------------------------------------------------------------------------
// Kernel 0: fp32 -> bf16 conversion (x and the four weight matrices)
// ---------------------------------------------------------------------------
__global__ void cvt_f32_bf16(const float* __restrict__ src, __bf16* __restrict__ dst, int n) {
  int i = (blockIdx.x * blockDim.x + threadIdx.x) * 4;
  if (i + 3 < n) {
    float4 v = *(const float4*)(src + i);
    dst[i + 0] = (__bf16)v.x;
    dst[i + 1] = (__bf16)v.y;
    dst[i + 2] = (__bf16)v.z;
    dst[i + 3] = (__bf16)v.w;
  }
}

// ---------------------------------------------------------------------------
// Kernel 1: C = A[4096,1024] @ W^T + bias.  One wave owns a 16x64 tile.
// mode 0: store bf16 in head layout   [B,H,S,HD]   (Q, K)
// mode 1: store bf16 in head-T layout [B,H,HD,S]   (V transposed for attn*V)
// mode 2: store fp32 row-major [M,E]               (final output projection)
// ---------------------------------------------------------------------------
__global__ __launch_bounds__(128) void gemm_xwT(const __bf16* __restrict__ A,
                                                const __bf16* __restrict__ W,
                                                const float* __restrict__ bias,
                                                void* __restrict__ dst, int mode) {
  const int lane = threadIdx.x & 31;
  const int g = blockIdx.x * 4 + (threadIdx.x >> 5);  // global wave id, 0..4095
  const int m0 = (g >> 4) * 16;                       // 256 M tiles
  const int n0 = (g & 15) * 64;                       // 16 N super-tiles (64 wide)

  v8f acc[4] = {};
  for (int k0 = 0; k0 < E_; k0 += 32) {
    v16bf a = frag_a_bf16(A, E_, m0, k0, lane);
#pragma unroll
    for (int t = 0; t < 4; ++t) {
      v16bf b = frag_b_rowmajorT(W, E_, n0 + 16 * t, k0, lane);
      acc[t] = wmma_bf16(a, b, acc[t]);
    }
  }

  const int r0 = (lane & 16) ? 8 : 0;
#pragma unroll
  for (int t = 0; t < 4; ++t) {
    const int col = n0 + 16 * t + (lane & 15);
    const float bv = bias ? bias[col] : 0.0f;
#pragma unroll
    for (int r = 0; r < 8; ++r) {
      const int row = m0 + r0 + r;
      const float v = acc[t][r] + bv;
      if (mode == 2) {
        ((float*)dst)[(size_t)row * E_ + col] = v;
      } else {
        const int b = row >> 10, s = row & (S_ - 1);
        const int h = col >> 6,  d = col & (HD_ - 1);
        if (mode == 0)
          ((__bf16*)dst)[(((size_t)(b * H_ + h)) * S_ + s) * HD_ + d] = (__bf16)v;
        else
          ((__bf16*)dst)[(((size_t)(b * H_ + h)) * HD_ + d) * S_ + s] = (__bf16)v;
      }
    }
  }
}

// ---------------------------------------------------------------------------
// Kernel 2: per-(b,h,s) squared norms of q and k  (for the RBF branch)
// ---------------------------------------------------------------------------
__global__ void head_norms(const __bf16* __restrict__ Q, const __bf16* __restrict__ K,
                           float* __restrict__ qn, float* __restrict__ kn) {
  const int i = blockIdx.x * blockDim.x + threadIdx.x;  // [0, BH*S)
  if (i >= BH_ * S_) return;
  const __bf16* q = Q + (size_t)i * HD_;
  const __bf16* k = K + (size_t)i * HD_;
  float sq = 0.f, sk = 0.f;
#pragma unroll
  for (int d = 0; d < HD_; ++d) {
    float a = (float)q[d]; sq += a * a;
    float b = (float)k[d]; sk += b * b;
  }
  qn[i] = sq;
  kn[i] = sk;
}

// ---------------------------------------------------------------------------
// Kernel 3: raw scores qk = Q K^T via WMMA. One block = 4 waves sharing one
// (b,h); wave w owns rows m0..m0+15. The 16x64 K-tile sequence is identical
// for all 4 waves, so it is staged through a double-buffered LDS tile with
// CDNA5 async-LDS DMA (each wave copies a 512B quarter per tile).
//
// Row statistics are accumulated per-lane (no cross-lane traffic in the hot
// loop): running max of s, sum of exp(s) (unshifted), and the RBF sum. A
// single 16-lane butterfly per statistic at the end produces the row values;
// softsum is stored max-shifted: sum(exp(s - max)) = exp(-max) * sum(exp(s)).
// ---------------------------------------------------------------------------
__global__ __launch_bounds__(128) void scores_kernel(
    const __bf16* __restrict__ Q, const __bf16* __restrict__ K,
    const float* __restrict__ qn, const float* __restrict__ kn,
    float* __restrict__ attn, float* __restrict__ rowmax,
    float* __restrict__ softsum, float* __restrict__ rbfsum) {
  const int lane = threadIdx.x & 31;
  const int wv   = threadIdx.x >> 5;
  const int bh   = blockIdx.x >> 4;                     // 64 (b,h) pairs
  const int m0   = ((blockIdx.x & 15) * 4 + wv) * 16;   // 64 row strips per bh

  const __bf16* Qh = Q + (size_t)bh * S_ * HD_;
  const __bf16* Kh = K + (size_t)bh * S_ * HD_;
  const float* qnh = qn + bh * S_;
  const float* knh = kn + bh * S_;
  float* att = attn + (size_t)bh * S_ * S_;

  // Q rows are reused across the whole j loop: load both K-halves once.
  v16bf a0 = frag_a_bf16(Qh, HD_, m0, 0, lane);
  v16bf a1 = frag_a_bf16(Qh, HD_, m0, 32, lane);

  const int r0 = (lane & 16) ? 8 : 0;
  float qr[8], sg[8], pmax[8], psum[8], prbf[8];
#pragma unroll
  for (int r = 0; r < 8; ++r) {
    qr[r] = qnh[m0 + r0 + r];
    sg[r] = fminf(fmaxf(qr[r], 1e-8f), 1e4f);  // sigma^2 = clip(q_norm)
    pmax[r] = -3.0e38f;
    psum[r] = 0.f;
    prbf[r] = 0.f;
  }

#if USE_ASYNC_LDS
  __shared__ alignas(64) __bf16 ktile[2][16 * HD_];  // 2 x 2KB double buffer
  const unsigned ldsBase = (unsigned)(size_t)(void*)&ktile[0][0];
  const unsigned myOff = (unsigned)((wv * 32 + lane) * 16);  // this lane's 16B slice
  // Prologue: stage tile 0 into buffer 0 (512B per wave, b128 per lane).
  async_ld_b128(ldsBase + myOff, (unsigned long long)(size_t)Kh + myOff);
#endif

  for (int t = 0; t < S_ / 16; ++t) {
    const int j0 = t * 16;
    v8f c = {};
#if USE_ASYNC_LDS
    const int buf = t & 1;
    if (t + 1 < S_ / 16) {  // stage next tile into the other buffer
      const unsigned dst = ldsBase + (unsigned)((buf ^ 1) * 16 * HD_ * 2) + myOff;
      const unsigned long long src =
          (unsigned long long)(size_t)(Kh + (size_t)(j0 + 16) * HD_) + myOff;
      async_ld_b128(dst, src);
      wait_async1();  // my older (current-tile) DMA has landed
    } else {
      wait_async0();
    }
    __syncthreads();  // all 4 waves' quarters of the current tile are in LDS
    {
      v16bf b0 = frag_b_lds(&ktile[buf][0], 0, lane);
      v16bf b1 = frag_b_lds(&ktile[buf][0], 32, lane);
      c = wmma_bf16(a0, b0, c);
      c = wmma_bf16(a1, b1, c);
    }
#else
    {
      v16bf b0 = frag_b_rowmajorT(Kh, HD_, j0, 0, lane);
      v16bf b1 = frag_b_rowmajorT(Kh, HD_, j0, 32, lane);
      c = wmma_bf16(a0, b0, c);
      c = wmma_bf16(a1, b1, c);
    }
#endif

    const float knl = knh[j0 + (lane & 15)];
#pragma unroll
    for (int r = 0; r < 8; ++r) {
      const float qk = c[r];
      // coalesced: lanes of one half write 16 contiguous floats per row
      att[(size_t)(m0 + r0 + r) * S_ + j0 + (lane & 15)] = qk;

      const float s = qk * 0.125f;  // 1/sqrt(HD)
      pmax[r] = fmaxf(pmax[r], s);
      psum[r] += __expf(s);

      const float d2 = fmaxf(qr[r] + knl - 2.f * qk, 0.f);
      prbf[r] += __expf(-d2 / sg[r]);
    }

#if USE_ASYNC_LDS
    __syncthreads();  // protect this buffer from being overwritten next iter
#endif
  }

  // One butterfly per statistic per row (lane halves reduce independently).
#pragma unroll
  for (int r = 0; r < 8; ++r) {
    pmax[r] = fmaxf(pmax[r], __shfl_xor(pmax[r], 1, 32));
    pmax[r] = fmaxf(pmax[r], __shfl_xor(pmax[r], 2, 32));
    pmax[r] = fmaxf(pmax[r], __shfl_xor(pmax[r], 4, 32));
    pmax[r] = fmaxf(pmax[r], __shfl_xor(pmax[r], 8, 32));
    psum[r] += __shfl_xor(psum[r], 1, 32);
    psum[r] += __shfl_xor(psum[r], 2, 32);
    psum[r] += __shfl_xor(psum[r], 4, 32);
    psum[r] += __shfl_xor(psum[r], 8, 32);
    prbf[r] += __shfl_xor(prbf[r], 1, 32);
    prbf[r] += __shfl_xor(prbf[r], 2, 32);
    prbf[r] += __shfl_xor(prbf[r], 4, 32);
    prbf[r] += __shfl_xor(prbf[r], 8, 32);
  }

  if ((lane & 15) == 0) {  // lane 0 owns rows m0+0..7, lane 16 owns m0+8..15
#pragma unroll
    for (int r = 0; r < 8; ++r) {
      const int row = bh * S_ + m0 + r0 + r;
      rowmax[row]  = pmax[r];
      softsum[row] = psum[r] * __expf(-pmax[r]);  // = sum(exp(s - max))
      rbfsum[row]  = prbf[r];
    }
  }
}

// ---------------------------------------------------------------------------
// Kernel 4: turn raw qk into final blended attention, in place (float4/lane)
// ---------------------------------------------------------------------------
__global__ void normalize_kernel(float* __restrict__ attn,
                                 const float* __restrict__ rowmax,
                                 const float* __restrict__ softsum,
                                 const float* __restrict__ rbfsum,
                                 const float* __restrict__ qn,
                                 const float* __restrict__ kn,
                                 const float* __restrict__ pond) {
  const size_t e0 = ((size_t)blockIdx.x * blockDim.x + threadIdx.x) * 4;
  const int bh  = (int)(e0 >> 20);           // S*S = 2^20
  const int rem = (int)(e0 & ((1u << 20) - 1));
  const int i = rem >> 10;
  const int j = rem & (S_ - 1);
  const int row = bh * S_ + i;

  const float rm = rowmax[row];
  const float ss = softsum[row];
  const float rs = fmaxf(rbfsum[row], 1e-8f);
  const float q  = qn[row];
  const float sig = fminf(fmaxf(q, 1e-8f), 1e4f);

  const float p  = pond[0];
  const float w1 = 1.f / (1.f + __expf(-p));
  const float w0 = 1.f - w1;
  const float inv = 1.f / (w0 + w1 + 1e-7f);

  float4 qk = *(const float4*)(attn + e0);
  const float* knh = kn + bh * S_ + j;
  float qkv[4] = {qk.x, qk.y, qk.z, qk.w};
  float out[4];
#pragma unroll
  for (int u = 0; u < 4; ++u) {
    const float soft = __expf(qkv[u] * 0.125f - rm) / ss;
    const float d2 = fmaxf(q + knh[u] - 2.f * qkv[u], 0.f);
    float rb = __expf(-d2 / sig) / rs;
    if (!__builtin_isfinite(rb)) rb = 0.f;  // nan_to_num
    out[u] = (w0 * soft + w1 * rb) * inv;
  }
  *(float4*)(attn + e0) = make_float4(out[0], out[1], out[2], out[3]);
}

// ---------------------------------------------------------------------------
// Kernel 5: vals = attn @ V via WMMA. A = attn fp32 (converted per-fragment),
// B = V^T bf16 [B,H,HD,S] so B fragments are contiguous in K. attn (256MB)
// exceeds L2, so prefetch this lane's row two K-chunks (64 floats) ahead.
// ---------------------------------------------------------------------------
__global__ __launch_bounds__(128) void attn_v_kernel(const float* __restrict__ attn,
                                                     const __bf16* __restrict__ Vt,
                                                     __bf16* __restrict__ vals) {
  const int lane = threadIdx.x & 31;
  const int g = blockIdx.x * 4 + (threadIdx.x >> 5);  // 0..4095
  const int bh = g >> 6;
  const int m0 = (g & 63) * 16;
  const float* att = attn + (size_t)bh * S_ * S_;
  const __bf16* Vh = Vt + (size_t)bh * HD_ * S_;

  v8f acc[4] = {};
  for (int k0 = 0; k0 < S_; k0 += 32) {
    if (k0 + 64 < S_)  // global_prefetch_b8 of my attn row, two chunks (64 floats) ahead
      __builtin_prefetch(att + (size_t)(m0 + (lane & 15)) * S_ + k0 + 64, 0, 1);
    v16bf a = frag_a_f32(att, S_, m0, k0, lane);
#pragma unroll
    for (int t = 0; t < 4; ++t) {
      v16bf b = frag_b_rowmajorT(Vh, S_, 16 * t, k0, lane);
      acc[t] = wmma_bf16(a, b, acc[t]);
    }
  }

  const int b = bh >> 4, h = bh & 15;
  const int r0 = (lane & 16) ? 8 : 0;
#pragma unroll
  for (int t = 0; t < 4; ++t) {
    const int col = h * HD_ + 16 * t + (lane & 15);
#pragma unroll
    for (int r = 0; r < 8; ++r) {
      const int s = m0 + r0 + r;
      vals[((size_t)(b * S_ + s)) * E_ + col] = (__bf16)acc[t][r];
    }
  }
}

// ---------------------------------------------------------------------------
// Host-side launch
// ---------------------------------------------------------------------------
extern "C" void kernel_launch(void* const* d_in, const int* in_sizes, int n_in,
                              void* d_out, int out_size, void* d_ws, size_t ws_size,
                              hipStream_t stream) {
  const float* x    = (const float*)d_in[0];
  const float* Wq   = (const float*)d_in[1];
  const float* bq   = (const float*)d_in[2];
  const float* Wk   = (const float*)d_in[3];
  const float* bk   = (const float*)d_in[4];
  const float* Wv   = (const float*)d_in[5];
  const float* bv   = (const float*)d_in[6];
  const float* Wo   = (const float*)d_in[7];
  const float* bo   = (const float*)d_in[8];
  const float* pond = (const float*)d_in[9];

  float* out  = (float*)d_out;                 // [B,S,E]
  float* attn = out + (size_t)M_ * E_;         // [B,H,S,S] (tuple element 2)

  // Workspace layout (~49.3 MB)
  char* w = (char*)d_ws;
  __bf16* xb    = (__bf16*)(w + 0);            // 8 MB  [M,E]
  __bf16* Wqb   = (__bf16*)(w + 8388608);      // 2 MB
  __bf16* Wkb   = (__bf16*)(w + 10485760);     // 2 MB
  __bf16* Wvb   = (__bf16*)(w + 12582912);     // 2 MB
  __bf16* Wob   = (__bf16*)(w + 14680064);     // 2 MB
  __bf16* Qb    = (__bf16*)(w + 16777216);     // 8 MB  [B,H,S,HD]
  __bf16* Kb    = (__bf16*)(w + 25165824);     // 8 MB  [B,H,S,HD]
  __bf16* Vtb   = (__bf16*)(w + 33554432);     // 8 MB  [B,H,HD,S]
  __bf16* valsb = (__bf16*)(w + 41943040);     // 8 MB  [B,S,E]
  float*  qn    = (float*)(w + 50331648);      // 256 KB [B,H,S]
  float*  kn    = (float*)(w + 50593792);      // 256 KB
  float*  rmax  = (float*)(w + 50855936);      // 256 KB
  float*  ssum  = (float*)(w + 51118080);      // 256 KB
  float*  rsum  = (float*)(w + 51380224);      // 256 KB

  // 0) fp32 -> bf16
  cvt_f32_bf16<<<4096, 256, 0, stream>>>(x,  xb,  M_ * E_);
  cvt_f32_bf16<<<1024, 256, 0, stream>>>(Wq, Wqb, E_ * E_);
  cvt_f32_bf16<<<1024, 256, 0, stream>>>(Wk, Wkb, E_ * E_);
  cvt_f32_bf16<<<1024, 256, 0, stream>>>(Wv, Wvb, E_ * E_);
  cvt_f32_bf16<<<1024, 256, 0, stream>>>(Wo, Wob, E_ * E_);

  // 1) QKV projections (WMMA)
  gemm_xwT<<<1024, 128, 0, stream>>>(xb, Wqb, bq, (void*)Qb, 0);
  gemm_xwT<<<1024, 128, 0, stream>>>(xb, Wkb, bk, (void*)Kb, 0);
  gemm_xwT<<<1024, 128, 0, stream>>>(xb, Wvb, bv, (void*)Vtb, 1);

  // 2) per-row norms for the RBF branch
  head_norms<<<256, 256, 0, stream>>>(Qb, Kb, qn, kn);

  // 3) raw QK^T (WMMA + async-LDS K-tile staging) + row statistics
  scores_kernel<<<1024, 128, 0, stream>>>(Qb, Kb, qn, kn, attn, rmax, ssum, rsum);

  // 4) blend softmax + RBF in place -> final attn output
  normalize_kernel<<<65536, 256, 0, stream>>>(attn, rmax, ssum, rsum, qn, kn, pond);

  // 5) attn @ V (WMMA)
  attn_v_kernel<<<1024, 128, 0, stream>>>(attn, Vtb, valsb);

  // 6) output projection (WMMA, fp32 store + bias)
  gemm_xwT<<<1024, 128, 0, stream>>>(valsb, Wob, bo, (void*)out, 2);
}
